// CompressRestorationModule_10084583211501
// MI455X (gfx1250) — matrix operand order repaired and checked by reading, not compiled, not run on hardware
//
#include <hip/hip_runtime.h>
#include <stdint.h>

typedef __attribute__((ext_vector_type(16))) _Float16 v16h;
typedef __attribute__((ext_vector_type(8)))  _Float16 v8h;
typedef __attribute__((ext_vector_type(8)))  float    v8f;

#define HDIM 256
#define WDIM 256
#define BATCH 4
#define NPIX (BATCH * HDIM * WDIM)

// LDS byte offset of a __shared__ object (flat LDS aperture keeps the offset
// in the low 32 bits; used as the VDST operand of async global->LDS loads).
__device__ __forceinline__ unsigned lds_off_u32(const void* p) {
    return (unsigned)(uintptr_t)p;
}

// ---------------------------------------------------------------------------
// Weight packing: f32 OIHW  ->  f16 [tap][kg][ntile][n(16)][k(32)]
// so each WMMA lane reads one contiguous 32B v16h (B-matrix 32x16 layout:
// lanes 0-15: N=lane, K=0..7,16..23; lanes 16-31: N=lane-16, K=8..15,24..31).
// ---------------------------------------------------------------------------
__global__ void pack_weights_kernel(const float* __restrict__ w,
                                    _Float16* __restrict__ wp,
                                    int cout, int cin, int kh)
{
    int idx = blockIdx.x * blockDim.x + threadIdx.x;
    int taps = kh * kh;
    int kgn  = cin >> 5;
    int ntn  = (cout + 15) >> 4;
    int total = taps * kgn * ntn * 16 * 32;
    if (idx >= total) return;
    int k = (((idx >> 3) & 1) << 4) | (((idx >> 4) & 1) << 3) | (idx & 7);
    int q = idx >> 5;
    int n = q & 15;  q >>= 4;
    int nt = q % ntn; q /= ntn;
    int kg = q % kgn;
    int tap = q / kgn;
    int o = nt * 16 + n;
    int c = kg * 32 + k;
    float v = (o < cout) ? w[((size_t)o * cin + c) * taps + tap] : 0.0f;
    wp[idx] = (_Float16)v;
}

// ---------------------------------------------------------------------------
// Small-Cin 3x3 conv (Cin = 1 or 2), pad=1, optional ReLU.
// f32 NCHW input(s) -> f16 NHWC (C=32) output.  VALU path (K too small for WMMA).
// ---------------------------------------------------------------------------
__global__ void conv_small_kernel(const float* __restrict__ in0,
                                  const float* __restrict__ in1,
                                  const float* __restrict__ w,     // [32][cin][3][3]
                                  const float* __restrict__ bias,  // [32]
                                  int cin, int relu,
                                  _Float16* __restrict__ out)      // NHWC, C=32
{
    int p = blockIdx.x * blockDim.x + threadIdx.x;
    if (p >= NPIX) return;
    int x = p & (WDIM - 1);
    int y = (p / WDIM) & (HDIM - 1);
    int b = p / (WDIM * HDIM);
    const float* ins[2] = { in0, in1 };
    float vin[2][9];
    for (int c = 0; c < cin; ++c)
        for (int t = 0; t < 9; ++t) {
            int yy = y + t / 3 - 1;
            int xx = x + t % 3 - 1;
            vin[c][t] = (yy >= 0 && yy < HDIM && xx >= 0 && xx < WDIM)
                      ? ins[c][((size_t)b * HDIM + yy) * WDIM + xx] : 0.0f;
        }
    for (int o = 0; o < 32; ++o) {
        float acc = bias[o];
        for (int c = 0; c < cin; ++c) {
#pragma unroll
            for (int t = 0; t < 9; ++t)
                acc += vin[c][t] * w[((size_t)o * cin + c) * 9 + t];
        }
        if (relu && acc < 0.0f) acc = 0.0f;
        out[(size_t)p * 32 + o] = (_Float16)acc;
    }
}

// ---------------------------------------------------------------------------
// b_inout = a - b_inout   (elementwise f16; used for fen - fbn)
// ---------------------------------------------------------------------------
__global__ void sub_kernel(const _Float16* __restrict__ a,
                           _Float16* __restrict__ b_inout, int n)
{
    int i = blockIdx.x * blockDim.x + threadIdx.x;
    if (i < n) b_inout[i] = (_Float16)((float)a[i] - (float)b_inout[i]);
}

// ---------------------------------------------------------------------------
// Generic WMMA implicit-GEMM conv.
//   KH: 1 or 3 (pad = (KH/2)*DIL), DIL: dilation, CIN32: Cin/32, NT: Cout tiles.
//   One wave handles a 16-pixel M-tile (contiguous in W) x 16-channel N-tile.
//   Packed weights are staged global->LDS once per block via CDNA5 async
//   loads (global_load_async_to_lds_b128, ASYNCcnt); B served by ds_load_b128.
//   A-matrix lane striping (16-bit 16x32): lane L -> M = L%16,
//   K = {hi*8..+7, 16+hi*8..+7} with hi = L>=16 -> two 16B global loads.
//   D (f32 16x16): lane L -> N = L%16, rows M = hi*8 + r.
// ---------------------------------------------------------------------------
template<int KH, int DIL, int CIN32, int NT, bool RELU, bool OUT_NCHW>
__global__ void __launch_bounds__(256)
conv_wmma_kernel(const _Float16* __restrict__ in, int in_cstride,
                 const _Float16* __restrict__ wp,
                 const float* __restrict__ bias, int cout_real,
                 _Float16* __restrict__ out16, float* __restrict__ out32,
                 int out_cstride, int out_choff)
{
    constexpr int WELEMS = KH * KH * CIN32 * NT * 512;  // packed f16 weights
    constexpr int WBYTES = WELEMS * 2;
    __shared__ _Float16 lw[WELEMS];

    // ---- async stage packed weights into LDS (16B per lane per issue) ----
    {
        const unsigned ldsbase = lds_off_u32(lw);
        const uint64_t gbase   = (uint64_t)wp;
        constexpr int ITERS = (WBYTES + 256 * 16 - 1) / (256 * 16);
#pragma unroll
        for (int it = 0; it < ITERS; ++it) {
            const unsigned off = (unsigned)(it * 256 + threadIdx.x) * 16u;
            if (off < (unsigned)WBYTES) {
                asm volatile("global_load_async_to_lds_b128 %0, %1, %2"
                             :
                             : "v"(ldsbase + off), "v"(off), "s"(gbase)
                             : "memory");
            }
        }
        asm volatile("s_wait_asynccnt 0x0" ::: "memory");
    }
    __syncthreads();

    const int MT   = NPIX / 16;
    const int wave = threadIdx.x >> 5;
    const int lane = threadIdx.x & 31;
    const int tile = blockIdx.x * 8 + wave;
    if (tile >= MT * NT) return;                 // wave-uniform exit (after barrier)
    const int mt    = tile / NT;
    const int nt    = tile - mt * NT;
    const int pbase = mt * 16;
    const int hi    = lane >> 4;
    const int mloc  = lane & 15;
    const int xbase = pbase & (WDIM - 1);        // 16-pixel tile stays in one row
    const int yrow  = (pbase / WDIM) & (HDIM - 1);
    const int bidx  = pbase / (WDIM * HDIM);
    const int lx    = xbase + mloc;

    v8f acc = {};
#pragma unroll
    for (int ky = 0; ky < KH; ++ky) {
        const int  yy  = yrow + (ky - KH / 2) * DIL;
        const bool yok = (yy >= 0) && (yy < HDIM);
#pragma unroll
        for (int kx = 0; kx < KH; ++kx) {
            const int  xx = lx + (kx - KH / 2) * DIL;
            const bool ok = yok && (xx >= 0) && (xx < WDIM);
            const _Float16* ppix =
                in + ((size_t)(bidx * HDIM + yy) * WDIM + xx) * in_cstride;
#pragma unroll
            for (int kg = 0; kg < CIN32; ++kg) {
                v16h amat = {};
                if (ok) {
                    const _Float16* pc = ppix + kg * 32 + hi * 8;
                    v8h alo = *(const v8h*)pc;          // K = hi*8 .. +7
                    v8h ahi = *(const v8h*)(pc + 16);   // K = 16+hi*8 .. +7
#pragma unroll
                    for (int i = 0; i < 8; ++i) { amat[i] = alo[i]; amat[8 + i] = ahi[i]; }
                }
                const int tap = ky * KH + kx;
                const _Float16* pb =
                    lw + (((size_t)(tap * CIN32 + kg) * NT + nt) * 16 + mloc) * 32 + hi * 16;
                v16h bmat = *(const v16h*)pb;           // ds_load_b128
                acc = __builtin_amdgcn_wmma_f32_16x16x32_f16(
                          false, amat, false, bmat, (short)0, acc, false, false);
            }
        }
    }

    const int   n  = nt * 16 + mloc;
    const float bv = (n < cout_real) ? bias[n] : 0.0f;
#pragma unroll
    for (int r = 0; r < 8; ++r) {
        float v = acc[r] + bv;
        if (RELU) v = v > 0.0f ? v : 0.0f;
        const int m = hi * 8 + r;
        const int p = pbase + m;
        if (OUT_NCHW) {
            out32[(((size_t)bidx * cout_real + n) * HDIM + yrow) * WDIM + (xbase + m)] = v;
        } else {
            out16[(size_t)p * out_cstride + out_choff + n] = (_Float16)v;
        }
    }
}

// ---------------------------------------------------------------------------
// DCNv2: each lane bilinear-samples its pixel's channels for 2 of the 4
// deformable groups (matching the A-matrix K striping), applies the sigmoid
// mask (hardware V_RCP_F32 -- no IEEE divide sequence in the hot loop), then
// per 3x3 tap performs two 16x16x32 WMMAs (N-tiles 0/1, A reused).
// Tap weights staged into LDS.  Epilogue adds bias + fb residual.
// ---------------------------------------------------------------------------
__global__ void __launch_bounds__(256)
dcn_wmma_kernel(const _Float16* __restrict__ x,    // NHWC 32 (fen - fbn)
                const _Float16* __restrict__ om,   // NHWC 112: oy[36] ox[36] m[36] pad
                const _Float16* __restrict__ fb,   // NHWC 32 (residual)
                const _Float16* __restrict__ wp,   // packed [tap][nt(2)][16][32]
                const float* __restrict__ bias,    // [32]
                _Float16* __restrict__ out,        // NHWC 96 concat
                int out_choff)
{
    constexpr int WELEMS = 9 * 2 * 16 * 32;            // 9216 f16
    __shared__ _Float16 lw[WELEMS];
    for (int i = threadIdx.x; i < WELEMS / 8; i += 256)
        ((v8h*)lw)[i] = ((const v8h*)wp)[i];           // b128 load + ds_store
    __syncthreads();

    const int MT   = NPIX / 16;
    const int wave = threadIdx.x >> 5;
    const int lane = threadIdx.x & 31;
    const int mt   = blockIdx.x * 8 + wave;
    if (mt >= MT) return;
    const int pbase = mt * 16;
    const int hi    = lane >> 4;
    const int mloc  = lane & 15;
    const int xbase = pbase & (WDIM - 1);
    const int yrow  = (pbase / WDIM) & (HDIM - 1);
    const int bidx  = pbase / (WDIM * HDIM);
    const int px    = xbase + mloc;
    const int p     = pbase + mloc;
    const _Float16* ompix = om + (size_t)p * 112;

    v8f acc0 = {};
    v8f acc1 = {};
    for (int kk = 0; kk < 9; ++kk) {
        const int ky = kk / 3, kx = kk % 3;
        v16h amat = {};
#pragma unroll
        for (int half = 0; half < 2; ++half) {
            const int g = hi + half * 2;           // hi=0 -> groups 0,2; hi=1 -> 1,3
            float offy = (float)ompix[g * 9 + kk];
            float offx = (float)ompix[36 + g * 9 + kk];
            float mraw = (float)ompix[72 + g * 9 + kk];
            float msk  = __builtin_amdgcn_rcpf(1.0f + __expf(-mraw));  // sigmoid via v_rcp_f32
            float py   = (float)yrow + (float)(ky - 1) + offy;
            float pxx  = (float)px   + (float)(kx - 1) + offx;
            float y0f  = floorf(py), x0f = floorf(pxx);
            float ly   = py - y0f,   lxw = pxx - x0f;
            int   y0   = (int)y0f,   x0  = (int)x0f;
            float wgt[4] = { (1.0f - ly) * (1.0f - lxw), (1.0f - ly) * lxw,
                             ly * (1.0f - lxw),          ly * lxw };
            float vacc[8] = {0,0,0,0,0,0,0,0};
#pragma unroll
            for (int cy = 0; cy < 2; ++cy) {
#pragma unroll
                for (int cx = 0; cx < 2; ++cx) {
                    int yy = y0 + cy, xx = x0 + cx;
                    if (yy >= 0 && yy < HDIM && xx >= 0 && xx < WDIM) {
                        const v8h* src = (const v8h*)
                            (x + ((size_t)(bidx * HDIM + yy) * WDIM + xx) * 32 + g * 8);
                        v8h v = *src;
                        float wv = wgt[cy * 2 + cx];
#pragma unroll
                        for (int i = 0; i < 8; ++i) vacc[i] += wv * (float)v[i];
                    }
                }
            }
#pragma unroll
            for (int i = 0; i < 8; ++i) amat[half * 8 + i] = (_Float16)(vacc[i] * msk);
        }
        const _Float16* pb0 = lw + (((size_t)kk * 2 + 0) * 16 + mloc) * 32 + hi * 16;
        const _Float16* pb1 = lw + (((size_t)kk * 2 + 1) * 16 + mloc) * 32 + hi * 16;
        v16h b0 = *(const v16h*)pb0;
        v16h b1 = *(const v16h*)pb1;
        acc0 = __builtin_amdgcn_wmma_f32_16x16x32_f16(false, amat, false, b0, (short)0, acc0, false, false);
        acc1 = __builtin_amdgcn_wmma_f32_16x16x32_f16(false, amat, false, b1, (short)0, acc1, false, false);
    }

#pragma unroll
    for (int ntile = 0; ntile < 2; ++ntile) {
        const int   n  = ntile * 16 + mloc;
        const float bv = bias[n];
        const v8f&  ac = ntile ? acc1 : acc0;
#pragma unroll
        for (int r = 0; r < 8; ++r) {
            const int m  = hi * 8 + r;
            const int pp = pbase + m;
            float v = ac[r] + bv + (float)fb[(size_t)pp * 32 + n];
            out[(size_t)pp * 96 + out_choff + n] = (_Float16)v;
        }
    }
}

// ---------------------------------------------------------------------------
// Host-side orchestration
// ---------------------------------------------------------------------------
extern "C" void kernel_launch(void* const* d_in, const int* in_sizes, int n_in,
                              void* d_out, int out_size, void* d_ws, size_t ws_size,
                              hipStream_t stream)
{
    (void)in_sizes; (void)n_in; (void)out_size; (void)ws_size;

    // ---- inputs (setup_inputs dict order, nested dicts flattened) ----
    const float* b_i    = (const float*)d_in[0];
    const float* b_i_n  = (const float*)d_in[1];
    const float* e_i_n  = (const float*)d_in[2];
    const float* fw1 = (const float*)d_in[3];  const float* fb1b = (const float*)d_in[4];
    const float* fw2 = (const float*)d_in[5];  const float* fb2b = (const float*)d_in[6];
    const float* fw3 = (const float*)d_in[7];  const float* fb3b = (const float*)d_in[8];
    const float* win = (const float*)d_in[9];  const float* binb = (const float*)d_in[10];
    const float* wd1 = (const float*)d_in[11]; const float* bd1  = (const float*)d_in[12];
    const float* wd2 = (const float*)d_in[13]; const float* bd2  = (const float*)d_in[14];
    const float* wd3 = (const float*)d_in[15]; const float* bd3  = (const float*)d_in[16];
    const float* wout= (const float*)d_in[17]; const float* bout = (const float*)d_in[18];
    const float* dow[3]  = { (const float*)d_in[19], (const float*)d_in[23], (const float*)d_in[27] };
    const float* dob[3]  = { (const float*)d_in[20], (const float*)d_in[24], (const float*)d_in[28] };
    const float* dw[3]   = { (const float*)d_in[21], (const float*)d_in[25], (const float*)d_in[29] };
    const float* db[3]   = { (const float*)d_in[22], (const float*)d_in[26], (const float*)d_in[30] };
    const float* fuse_w  = (const float*)d_in[31];
    const float* fuse_b  = (const float*)d_in[32];
    float* out_f32 = (float*)d_out;

    // ---- workspace carve-up (f16 activations, channels-last) ----
    char* ws = (char*)d_ws;
    auto alloc = [&](size_t bytes) -> void* {
        void* pp = (void*)ws;
        ws += (bytes + 255) & ~(size_t)255;
        return pp;
    };
    _Float16* o16   = (_Float16*)alloc((size_t)NPIX * 32 * 2);
    _Float16* cat96 = (_Float16*)alloc((size_t)NPIX * 96 * 2);   // reused as lcat
    _Float16* feat  = (_Float16*)alloc((size_t)NPIX * 32 * 2);
    _Float16* fbuf1 = (_Float16*)alloc((size_t)NPIX * 32 * 2);
    _Float16* fbuf2 = (_Float16*)alloc((size_t)NPIX * 32 * 2);
    _Float16* fbuf3 = (_Float16*)alloc((size_t)NPIX * 32 * 2);
    _Float16* A1    = (_Float16*)alloc((size_t)NPIX * 32 * 2);
    _Float16* A2    = (_Float16*)alloc((size_t)NPIX * 32 * 2);
    _Float16* A3    = (_Float16*)alloc((size_t)NPIX * 32 * 2);
    _Float16* om16  = (_Float16*)alloc((size_t)NPIX * 112 * 2);
    _Float16* Ea    = o16;     // free after dilated convs
    _Float16* Eb    = cat96;   // free between feat and lcat phases

    // packed weights
    auto allocW = [&](int cout, int cin, int kh) -> _Float16* {
        size_t total = (size_t)kh * kh * (cin / 32) * ((cout + 15) / 16) * 512;
        return (_Float16*)alloc(total * 2);
    };
    _Float16* p_w2   = allocW(32, 32, 3);
    _Float16* p_w3   = allocW(32, 32, 3);
    _Float16* p_wd1  = allocW(32, 32, 3);
    _Float16* p_wd2  = allocW(32, 32, 3);
    _Float16* p_wd3  = allocW(32, 32, 3);
    _Float16* p_wout = allocW(32, 96, 1);
    _Float16* p_ow[3]= { allocW(108, 32, 3), allocW(108, 32, 3), allocW(108, 32, 3) };
    _Float16* p_dw[3]= { allocW(32, 32, 3), allocW(32, 32, 3), allocW(32, 32, 3) };
    _Float16* p_fuse = allocW(32, 96, 1);

    auto pack = [&](const float* w, _Float16* wp, int cout, int cin, int kh) {
        int total = kh * kh * (cin / 32) * ((cout + 15) / 16) * 512;
        pack_weights_kernel<<<(total + 255) / 256, 256, 0, stream>>>(w, wp, cout, cin, kh);
    };
    pack(fw2, p_w2, 32, 32, 3);   pack(fw3, p_w3, 32, 32, 3);
    pack(wd1, p_wd1, 32, 32, 3);  pack(wd2, p_wd2, 32, 32, 3);  pack(wd3, p_wd3, 32, 32, 3);
    pack(wout, p_wout, 32, 96, 1);
    for (int s = 0; s < 3; ++s) { pack(dow[s], p_ow[s], 108, 32, 3); pack(dw[s], p_dw[s], 32, 32, 3); }
    pack(fuse_w, p_fuse, 32, 96, 1);

    const int MT = NPIX / 16;
    const int blkSmall = (NPIX + 255) / 256;
    const int blk2  = MT * 2 / 8;   // NT=2 conv tiles (exact)
    const int blk7  = MT * 7 / 8;   // NT=7 conv tiles (exact)
    const int blkD  = MT / 8;       // DCN tiles (exact)
    const int nElem = NPIX * 32;

    // ---- MFEO offset head ----
    conv_small_kernel<<<blkSmall, 256, 0, stream>>>(b_i, b_i_n, win, binb, 2, 1, o16);
    conv_wmma_kernel<3,1,1,2,false,false><<<blk2, 256, 0, stream>>>(o16, 32, p_wd1, bd1, 32, cat96, nullptr, 96, 0);
    conv_wmma_kernel<3,2,1,2,false,false><<<blk2, 256, 0, stream>>>(o16, 32, p_wd2, bd2, 32, cat96, nullptr, 96, 32);
    conv_wmma_kernel<3,4,1,2,false,false><<<blk2, 256, 0, stream>>>(o16, 32, p_wd3, bd3, 32, cat96, nullptr, 96, 64);
    conv_wmma_kernel<1,1,3,2,false,false><<<blk2, 256, 0, stream>>>(cat96, 96, p_wout, bout, 32, feat, nullptr, 32, 0);

    // ---- FEM chains ----
    conv_small_kernel<<<blkSmall, 256, 0, stream>>>(b_i, nullptr, fw1, fb1b, 1, 1, fbuf1);
    conv_wmma_kernel<3,1,1,2,true ,false><<<blk2, 256, 0, stream>>>(fbuf1, 32, p_w2, fb2b, 32, fbuf2, nullptr, 32, 0);
    conv_wmma_kernel<3,1,1,2,false,false><<<blk2, 256, 0, stream>>>(fbuf2, 32, p_w3, fb3b, 32, fbuf3, nullptr, 32, 0);

    conv_small_kernel<<<blkSmall, 256, 0, stream>>>(b_i_n, nullptr, fw1, fb1b, 1, 1, A1);
    conv_wmma_kernel<3,1,1,2,true ,false><<<blk2, 256, 0, stream>>>(A1, 32, p_w2, fb2b, 32, A2, nullptr, 32, 0);
    conv_wmma_kernel<3,1,1,2,false,false><<<blk2, 256, 0, stream>>>(A2, 32, p_w3, fb3b, 32, A3, nullptr, 32, 0);

    conv_small_kernel<<<blkSmall, 256, 0, stream>>>(e_i_n, nullptr, fw1, fb1b, 1, 1, Ea);
    sub_kernel<<<nElem / 256, 256, 0, stream>>>(Ea, A1, nElem);                      // A1 = fen1 - fbn1
    conv_wmma_kernel<3,1,1,2,true ,false><<<blk2, 256, 0, stream>>>(Ea, 32, p_w2, fb2b, 32, Eb, nullptr, 32, 0);
    sub_kernel<<<nElem / 256, 256, 0, stream>>>(Eb, A2, nElem);                      // A2 = fen2 - fbn2
    conv_wmma_kernel<3,1,1,2,false,false><<<blk2, 256, 0, stream>>>(Eb, 32, p_w3, fb3b, 32, Ea, nullptr, 32, 0);
    sub_kernel<<<nElem / 256, 256, 0, stream>>>(Ea, A3, nElem);                      // A3 = fen3 - fbn3

    // ---- DCN x3 (om buffer reused; lcat aliases cat96) ----
    _Float16* diffs[3] = { A1, A2, A3 };
    _Float16* fbs[3]   = { fbuf1, fbuf2, fbuf3 };
    for (int s = 0; s < 3; ++s) {
        conv_wmma_kernel<3,1,1,7,false,false><<<blk7, 256, 0, stream>>>(
            feat, 32, p_ow[s], dob[s], 108, om16, nullptr, 112, 0);
        dcn_wmma_kernel<<<blkD, 256, 0, stream>>>(
            diffs[s], om16, fbs[s], p_dw[s], db[s], cat96, 32 * s);
    }

    // ---- final 1x1 fuse: 96 -> 32, f32 NCHW into d_out ----
    conv_wmma_kernel<1,1,3,2,false,true><<<blk2, 256, 0, stream>>>(
        cat96, 96, p_fuse, fuse_b, 32, nullptr, out_f32, 0, 0);
}
